// Correlation_29618094474071
// MI455X (gfx1250) — matrix operand order repaired and checked by reading, not compiled, not run on hardware
//
#include <hip/hip_runtime.h>
#include <hip/hip_bf16.h>

// Shapes fixed by the reference: x [16, 256, 2048] fp32 -> out [16, 2048, 2048] fp32
#define NB 16
#define NC 256
#define NP 2048

// LDS B-tile row pitch in bf16 elements: 32 K-values + 8 pad -> 80 B rows.
// Makes b128 LDS loads/stores bank-conflict-free (lane banks hit 0,4,8,...,60).
#define BPITCH 40

typedef __attribute__((ext_vector_type(8)))  __bf16 v8bf;
typedef __attribute__((ext_vector_type(16))) __bf16 v16bf;
typedef __attribute__((ext_vector_type(8)))  float  v8f;

union FragU { v16bf f; v8bf h[2]; };

// gfx1250 async memory->LDS path (ASYNCcnt), if the toolchain exposes it.
#if defined(__has_builtin)
#if __has_builtin(__builtin_amdgcn_global_load_async_to_lds_b128) && \
    __has_builtin(__builtin_amdgcn_s_wait_asynccnt)
#define USE_ASYNC_LDS 1
#endif
#endif
#ifndef USE_ASYNC_LDS
#define USE_ASYNC_LDS 0
#endif

// Parameter types per clang diagnostic: b128 variant takes
// 'int __attribute__((vector_size(16)))' pointers (AS1 src, AS3 dst).
typedef int vint4 __attribute__((vector_size(16)));
typedef __attribute__((address_space(1))) vint4 gvint4;   // global (__device__)
typedef __attribute__((address_space(3))) vint4 lvint4;   // LDS (__shared__)

// ---------------------------------------------------------------------------
// Pass 1: per (b,p) column: norm, fold 1/norm into data, split fp32 -> bf16
// hi/lo, store transposed y[b][p][c] (c contiguous) so GEMM fragments are
// per-lane contiguous b128 loads.
// ---------------------------------------------------------------------------
__global__ __launch_bounds__(256)
void corr_prep_kernel(const float* __restrict__ x,
                      __bf16* __restrict__ yhi,
                      __bf16* __restrict__ ylo) {
  int gid = blockIdx.x * 256 + threadIdx.x;       // 0 .. NB*NP-1
  const float* xp = x + ((size_t)(gid >> 11)) * NC * NP + (gid & (NP - 1));

  float sum = 0.f;
#pragma unroll 8
  for (int c = 0; c < NC; ++c) {
    float v = xp[(size_t)c * NP];                 // coalesced across threads
    sum = fmaf(v, v, sum);
  }
  // reference: denom = max(np*nq, 1e-8); fold per-point as 1/max(np, 1e-4)
  float inv = 1.0f / fmaxf(sqrtf(sum), 1e-4f);

  __bf16* hi = yhi + (size_t)gid * NC;
  __bf16* lo = ylo + (size_t)gid * NC;

  for (int c0 = 0; c0 < NC; c0 += 8) {
    v8bf h, l;
#pragma unroll
    for (int j = 0; j < 8; ++j) {
      float v  = xp[(size_t)(c0 + j) * NP] * inv;
      __bf16 vh = (__bf16)v;                      // RNE
      h[j] = vh;
      l[j] = (__bf16)(v - (float)vh);             // residual
    }
    *(v8bf*)(hi + c0) = h;                        // global_store_b128
    *(v8bf*)(lo + c0) = l;
  }
}

// A fragment, 16-bit A-matrix 16x32 (MxK), lane = 16*half + m:
//   elements 0..7  = K = k0 + 8*half + (0..7)
//   elements 8..15 = K = k0 + 16 + 8*half + (0..7)
__device__ inline v16bf load_frag_a(const __bf16* __restrict__ row, int k0, int half) {
  FragU u;
  u.h[0] = *(const v8bf*)(row + k0 + 8 * half);
  u.h[1] = *(const v8bf*)(row + k0 + 16 + 8 * half);
  return u.f;
}

// B fragment from LDS tile, 16-bit B-matrix 32x16 (KxN), lane = 16*half + n:
//   elements 0..7  = K = 8*half + (0..7)
//   elements 8..15 = K = 16 + 8*half + (0..7)
__device__ inline v16bf load_frag_b_lds(const __bf16* row, int half) {
  FragU u;
  u.h[0] = *(const v8bf*)(row + 8 * half);        // ds_load_b128
  u.h[1] = *(const v8bf*)(row + 16 + 8 * half);
  return u.f;
}

// ---------------------------------------------------------------------------
// Pass 2: sim = yhat^T yhat via bf16 WMMA with hi/lo error compensation:
//   dot ~= hi*hi + hi*lo + lo*hi   (lo*lo ~ 2^-32, dropped)
// Block = 256 threads = 8 waves sharing one 64-wide q strip. The 64x32 B tile
// (hi+lo) is staged into double-buffered LDS with GLOBAL_LOAD_ASYNC_TO_LDS
// (no VGPR round trip), overlapping the copy of chunk k+1 with the 12 WMMAs
// of chunk k. Each wave computes a 16(p) x 64(q) strip.
// ---------------------------------------------------------------------------
__global__ __launch_bounds__(256)
void corr_gemm_kernel(const __bf16* __restrict__ yhi,
                      const __bf16* __restrict__ ylo,
                      float* __restrict__ out) {
  __shared__ __bf16 sH[2][64 * BPITCH];           // 2 x 5 KB, padded rows
  __shared__ __bf16 sL[2][64 * BPITCH];           // 2 x 5 KB

  const int lane = threadIdx.x & 31;
  const int wave = threadIdx.x >> 5;              // 0..7
  const int m    = lane & 15;
  const int half = lane >> 4;

  const int b     = blockIdx.z;
  const int pBase = blockIdx.y * 128 + wave * 16; // 8 waves tile p by 16
  const int qBase = blockIdx.x * 64;              // 4 q-subtiles of 16

  const __bf16* hbase = yhi + (size_t)b * NP * NC;
  const __bf16* lbase = ylo + (size_t)b * NP * NC;

  const __bf16* aH = hbase + (size_t)(pBase + m) * NC;
  const __bf16* aL = lbase + (size_t)(pBase + m) * NC;

  // Cooperative B-tile staging coordinates: thread -> (row 0..63, quarter 0..3)
  const int srow = threadIdx.x >> 2;              // q row within tile
  const int squd = threadIdx.x & 3;               // 8-element K quarter
  const __bf16* gBH = hbase + (size_t)(qBase + srow) * NC + 8 * squd;
  const __bf16* gBL = lbase + (size_t)(qBase + srow) * NC + 8 * squd;
  const int sOff = srow * BPITCH + 8 * squd;

  auto stage = [&](int buf, int k0) {
#if USE_ASYNC_LDS
    __builtin_amdgcn_global_load_async_to_lds_b128(
        (gvint4*)(unsigned long long)(gBH + k0),
        (lvint4*)(unsigned int)(size_t)(&sH[buf][sOff]), 0, 0);
    __builtin_amdgcn_global_load_async_to_lds_b128(
        (gvint4*)(unsigned long long)(gBL + k0),
        (lvint4*)(unsigned int)(size_t)(&sL[buf][sOff]), 0, 0);
#else
    *(v8bf*)(&sH[buf][sOff]) = *(const v8bf*)(gBH + k0);
    *(v8bf*)(&sL[buf][sOff]) = *(const v8bf*)(gBL + k0);
#endif
  };

  v8f acc[4] = {};

  stage(0, 0);                                    // prologue copy into buffer 0

  for (int i = 0; i < 8; ++i) {                   // K chunks of 32
    const int k0  = i * 32;
    const int cur = i & 1;

    // A fragments from global (unique rows per wave, no sharing possible)
    v16bf ahi = load_frag_a(aH, k0, half);
    v16bf alo = load_frag_a(aL, k0, half);

#if USE_ASYNC_LDS
    __builtin_amdgcn_s_wait_asynccnt(0);          // my copies into buf[cur] done
#endif
    __syncthreads();                              // everyone's copies done;
                                                  // everyone done reading buf[cur^1]
    if (i < 7) stage(cur ^ 1, k0 + 32);           // overlap next copy with WMMAs

#pragma unroll
    for (int t = 0; t < 4; ++t) {
      const __bf16* rowH = &sH[cur][(t * 16 + m) * BPITCH];
      const __bf16* rowL = &sL[cur][(t * 16 + m) * BPITCH];
      v16bf bhi = load_frag_b_lds(rowH, half);
      v16bf blo = load_frag_b_lds(rowL, half);
      acc[t] = __builtin_amdgcn_wmma_f32_16x16x32_bf16(
          false, ahi, false, bhi, (short)0, acc[t], false, false);
      acc[t] = __builtin_amdgcn_wmma_f32_16x16x32_bf16(
          false, ahi, false, blo, (short)0, acc[t], false, false);
      acc[t] = __builtin_amdgcn_wmma_f32_16x16x32_bf16(
          false, alo, false, bhi, (short)0, acc[t], false, false);
    }
  }

  // Epilogue: C/D layout -> lane(0-15): N=lane, M=r; lane(16-31): N=lane-16, M=r+8
#pragma unroll
  for (int t = 0; t < 4; ++t) {
    const int q = qBase + t * 16 + m;
#pragma unroll
    for (int r = 0; r < 8; ++r) {
      const int p = pBase + r + 8 * half;
      float v = (p == q) ? 1.0f : acc[t][r];
      out[((size_t)b * NP + p) * NP + q] = v;     // q contiguous -> coalesced
    }
  }
}

extern "C" void kernel_launch(void* const* d_in, const int* in_sizes, int n_in,
                              void* d_out, int out_size, void* d_ws, size_t ws_size,
                              hipStream_t stream) {
  const float* x = (const float*)d_in[0];
  float* out = (float*)d_out;

  const size_t nElem = (size_t)NB * NP * NC;      // 8,388,608 elements
  __bf16* yhi = (__bf16*)d_ws;                    // 16 MB
  __bf16* ylo = (__bf16*)((char*)d_ws + nElem * sizeof(__bf16)); // +16 MB

  // Pass 1: normalize + bf16 hi/lo split, transposed layout
  corr_prep_kernel<<<(NB * NP) / 256, 256, 0, stream>>>(x, yhi, ylo);

  // Pass 2: WMMA Gram matrix with async-LDS double-buffered B tiles
  dim3 grid(NP / 64, NP / 128, NB);               // (32, 16, 16)
  corr_gemm_kernel<<<grid, 256, 0, stream>>>(yhi, ylo, out);
}
